// Attention_54262616817926
// MI455X (gfx1250) — matrix operand order repaired and checked by reading, not compiled, not run on hardware
//
#include <hip/hip_runtime.h>

typedef __attribute__((ext_vector_type(16))) __bf16          v16bf;
typedef __attribute__((ext_vector_type(8)))  float           v8f;
typedef __attribute__((ext_vector_type(16))) unsigned short  v16u;
typedef __attribute__((ext_vector_type(8)))  unsigned short  v8u;
typedef __attribute__((ext_vector_type(4)))  unsigned short  v4u;

#define ATTN_SCALE 0.14433756729740643f   // 48^-0.5

static __device__ __forceinline__ unsigned short f2bf(float f) {
  unsigned int u = __builtin_bit_cast(unsigned int, f);
  u += 0x7FFFu + ((u >> 16) & 1u);               // round-to-nearest-even
  return (unsigned short)(u >> 16);
}

// 16-half fragment from two 16B-aligned 8-half chunks (b128 x2, global or LDS)
static __device__ __forceinline__ v16bf frag16(const unsigned short* p0,
                                               const unsigned short* p1) {
  v8u a = *(const v8u*)p0;
  v8u b = *(const v8u*)p1;
  v16u u;
#pragma unroll
  for (int i = 0; i < 8; ++i) { u[i] = a[i]; u[i + 8] = b[i]; }
  return __builtin_bit_cast(v16bf, u);
}

static __device__ __forceinline__ v8f wmma_bf16(v16bf a, v16bf b, v8f c) {
  return __builtin_amdgcn_wmma_f32_16x16x32_bf16(false, a, false, b,
                                                 (short)0, c, false, false);
}

// gfx1250 async global->LDS copy, 16B per lane, ASYNCcnt-tracked
static __device__ __forceinline__ void async_b128(unsigned lds_off,
                                                  const unsigned short* g) {
  asm volatile("global_load_async_to_lds_b128 %0, %1, off"
               :: "v"(lds_off), "v"((unsigned long long)(size_t)g)
               : "memory");
}

// ---------------------------------------------------------------- fp32->bf16
__global__ void cvt4_kernel(const float* __restrict__ src,
                            unsigned short* __restrict__ dst, int n4) {
  int i = blockIdx.x * 256 + threadIdx.x;
  if (i < n4) {
    float4 f = ((const float4*)src)[i];
    v4u p;
    p[0] = f2bf(f.x); p[1] = f2bf(f.y); p[2] = f2bf(f.z); p[3] = f2bf(f.w);
    ((v4u*)dst)[i] = p;
  }
}

// ------------------------------------------------------------------ QKV GEMM
// grid (2048, 3): blockIdx.x = bp*64 + token_tile ; blockIdx.y = {Q,K,V}
// block 128: one wave per head. Weight fragments software-pipelined one
// K-step ahead so WMMA issue is not blocked by s_wait_loadcnt 0.
__global__ __launch_bounds__(128) void qkv_kernel(
    const unsigned short* __restrict__ xbf,   // [32][1024][192]
    const unsigned short* __restrict__ wbf,   // [576][192]
    const float*          __restrict__ bqkv,  // [576]
    unsigned short* __restrict__ Q,           // [32][4][1024][64]
    unsigned short* __restrict__ K,           // [32][4][1024][64]
    unsigned short* __restrict__ Vt)          // [32][4][64][1024]
{
  const int lane = threadIdx.x & 31;
  const int head = threadIdx.x >> 5;
  const int mat  = blockIdx.y;                // 0=Q 1=K 2=V
  const int bp   = blockIdx.x >> 6;
  const int t0   = (blockIdx.x & 63) * 16;
  const int hi   = lane >> 4;
  const int l16  = lane & 15;

  const unsigned short* xrow = xbf + (size_t)bp * 1024 * 192;

  if (mat < 2) {                              // ---- Q / K ----
    const int fbase = mat * 192 + head * 48;
    v16bf bf[6];                              // whole x row as B-fragments
#pragma unroll
    for (int kk = 0; kk < 6; ++kk) {
      const unsigned short* p = xrow + (size_t)(t0 + l16) * 192 + kk * 32 + hi * 8;
      bf[kk] = frag16(p, p + 16);
    }
    v8f acc[3] = {};
    v16bf a[3];
#pragma unroll
    for (int ft = 0; ft < 3; ++ft) {          // prime K-step 0
      const unsigned short* ap =
          wbf + (size_t)(fbase + ft * 16 + l16) * 192 + hi * 8;
      a[ft] = frag16(ap, ap + 16);
    }
#pragma unroll
    for (int kk = 0; kk < 6; ++kk) {
      v16bf an[3];
      if (kk + 1 < 6) {                       // prefetch next weights
        const int k1 = (kk + 1) * 32;
#pragma unroll
        for (int ft = 0; ft < 3; ++ft) {
          const unsigned short* ap =
              wbf + (size_t)(fbase + ft * 16 + l16) * 192 + k1 + hi * 8;
          an[ft] = frag16(ap, ap + 16);
        }
      }
#pragma unroll
      for (int ft = 0; ft < 3; ++ft) acc[ft] = wmma_bf16(a[ft], bf[kk], acc[ft]);
      if (kk + 1 < 6) {
#pragma unroll
        for (int ft = 0; ft < 3; ++ft) a[ft] = an[ft];
      }
    }
    unsigned short* Ob = (mat == 0 ? Q : K) + ((size_t)bp * 4 + head) * 1024 * 64;
    const int token = t0 + l16;
#pragma unroll
    for (int ft = 0; ft < 3; ++ft) {
      const int fcol = ft * 16 + hi * 8;
      v8u pk;
#pragma unroll
      for (int v = 0; v < 8; ++v)
        pk[v] = f2bf(acc[ft][v] + bqkv[fbase + fcol + v]);
      *(v8u*)(Ob + (size_t)token * 64 + fcol) = pk;
    }
    v8u z = (v8u)0;                           // zero pad cols 48..63
    *(v8u*)(Ob + (size_t)token * 64 + 48 + hi * 8) = z;
  } else {                                    // ---- V (transposed out) ----
    const int fbase = 384 + head * 48;
    v16bf af[6];                              // whole x row as A-fragments
#pragma unroll
    for (int kk = 0; kk < 6; ++kk) {
      const unsigned short* p = xrow + (size_t)(t0 + l16) * 192 + kk * 32 + hi * 16;
      af[kk] = frag16(p, p + 8);
    }
    v8f acc[3] = {};
    v16bf b[3];
#pragma unroll
    for (int ft = 0; ft < 3; ++ft) {          // prime K-step 0
      const unsigned short* bpp =
          wbf + (size_t)(fbase + ft * 16 + l16) * 192 + hi * 16;
      b[ft] = frag16(bpp, bpp + 8);
    }
#pragma unroll
    for (int kk = 0; kk < 6; ++kk) {
      v16bf bn[3];
      if (kk + 1 < 6) {
        const int k1 = (kk + 1) * 32;
#pragma unroll
        for (int ft = 0; ft < 3; ++ft) {
          const unsigned short* bpp =
              wbf + (size_t)(fbase + ft * 16 + l16) * 192 + k1 + hi * 16;
          bn[ft] = frag16(bpp, bpp + 8);
        }
      }
#pragma unroll
      for (int ft = 0; ft < 3; ++ft) acc[ft] = wmma_bf16(af[kk], b[ft], acc[ft]);
      if (kk + 1 < 6) {
#pragma unroll
        for (int ft = 0; ft < 3; ++ft) b[ft] = bn[ft];
      }
    }
    unsigned short* Vb = Vt + ((size_t)bp * 4 + head) * 64 * 1024;
#pragma unroll
    for (int ft = 0; ft < 3; ++ft) {
      const int dcol = ft * 16 + l16;
      const float bias = bqkv[fbase + dcol];
      v8u pk;
#pragma unroll
      for (int v = 0; v < 8; ++v) pk[v] = f2bf(acc[ft][v] + bias);
      *(v8u*)(Vb + (size_t)dcol * 1024 + t0 + hi * 8) = pk;
    }
  }
}

// ----------------------------------------------------------- flash attention
// 8 waves/block share (bp,h): K/V^T tiles triple-buffered in LDS via
// global_load_async_to_lds_b128 (+ s_wait_asynccnt), one 32-key step ahead,
// single barrier per step (slot k+1 was last read at step k-2).
__global__ __launch_bounds__(256) void attn_kernel(
    const unsigned short* __restrict__ Q,     // [32][4][1024][64]
    const unsigned short* __restrict__ K,
    const unsigned short* __restrict__ Vt,    // [32][4][64][1024]
    unsigned short* __restrict__ O)           // [32][1024][192] bf16
{
  __shared__ __align__(16) unsigned short sK[3][32 * 64];  // [key][d]
  __shared__ __align__(16) unsigned short sV[3][64 * 32];  // [d][key]

  const int tid  = threadIdx.x;
  const int lane = tid & 31;
  const int wv   = tid >> 5;
  const int blk  = blockIdx.x;
  const int q0   = ((blk & 7) * 8 + wv) * 16;
  const int h    = (blk >> 3) & 3;
  const int bp   = blk >> 5;
  const int hi   = lane >> 4;
  const int l16  = lane & 15;

  const unsigned short* Qb = Q  + ((size_t)bp * 4 + h) * 1024 * 64;
  const unsigned short* Kb = K  + ((size_t)bp * 4 + h) * 1024 * 64;
  const unsigned short* Vb = Vt + ((size_t)bp * 4 + h) * 64 * 1024;

  const unsigned skoff = (unsigned)(size_t)&sK[0][0];
  const unsigned svoff = (unsigned)(size_t)&sV[0][0];

  // one b128 per thread per tile: K tile 32x64, V tile 64x32 (rows 48..63 unused)
  const unsigned short* gK0 = Kb + (size_t)(tid >> 3) * 64 + (tid & 7) * 8;
  const unsigned short* gV0 = Vb + (size_t)(tid >> 2) * 1024 + (tid & 3) * 8;
  const unsigned ldst = (unsigned)(tid * 16);

  // prologue: stage step 0 into slot 0
  async_b128(skoff + ldst, gK0);
  async_b128(svoff + ldst, gV0);

  v16bf bq[2];                                // Q^T B-fragments, reused 32x
#pragma unroll
  for (int dh = 0; dh < 2; ++dh) {
    const unsigned short* p = Qb + (size_t)(q0 + l16) * 64 + dh * 32 + hi * 8;
    bq[dh] = frag16(p, p + 16);
  }

  float mrun = -1e30f, lrun = 0.f;
  v8f zacc[3] = {};
  int buf = 0, bufn = 1;

  for (int kt = 0; kt < 32; ++kt) {
    if (kt + 1 < 32) {                        // prefetch next tile pair
      const int kn = (kt + 1) * 32;
      async_b128(skoff + (unsigned)bufn * (32 * 64 * 2) + ldst,
                 gK0 + (size_t)kn * 64);
      async_b128(svoff + (unsigned)bufn * (64 * 32 * 2) + ldst, gV0 + kn);
      asm volatile("s_wait_asynccnt 0x2" ::: "memory");   // slot kt landed
    } else {
      asm volatile("s_wait_asynccnt 0x0" ::: "memory");
    }
    __syncthreads();                          // all waves' copies visible

    const unsigned short* kb = &sK[buf][0];
    const unsigned short* vb = &sV[buf][0];

    v8f s[2] = {};
#pragma unroll
    for (int st = 0; st < 2; ++st) {          // S^T = K * Q^T (per-lane query)
      const int kr = st * 16 + l16;
#pragma unroll
      for (int dh = 0; dh < 2; ++dh) {
        const unsigned short* p = kb + kr * 64 + dh * 32 + hi * 8;
        s[st] = wmma_bf16(frag16(p, p + 16), bq[dh], s[st]);
      }
    }
    float mloc = -1e30f;
#pragma unroll
    for (int st = 0; st < 2; ++st)
#pragma unroll
      for (int v = 0; v < 8; ++v) {
        s[st][v] *= ATTN_SCALE;
        mloc = fmaxf(mloc, s[st][v]);
      }
    mloc = fmaxf(mloc, __shfl_xor(mloc, 16, 32));
    const float mnew  = fmaxf(mrun, mloc);
    const float alpha = __expf(mrun - mnew);
    float p0[8], p1[8], psum = 0.f;
#pragma unroll
    for (int v = 0; v < 8; ++v) {
      p0[v] = __expf(s[0][v] - mnew);
      p1[v] = __expf(s[1][v] - mnew);
      psum += p0[v] + p1[v];
    }
    psum += __shfl_xor(psum, 16, 32);
    lrun = lrun * alpha + psum;
    mrun = mnew;
#pragma unroll
    for (int dt = 0; dt < 3; ++dt)
#pragma unroll
      for (int v = 0; v < 8; ++v) zacc[dt][v] *= alpha;

    // assemble P^T B-fragment: half-swap via shfl_xor(16), pack to bf16
    v16u u;
#pragma unroll
    for (int v = 0; v < 8; ++v) {
      const float o0 = __shfl_xor(p0[v], 16, 32);
      const float o1 = __shfl_xor(p1[v], 16, 32);
      u[v]     = f2bf(hi ? o1 : p0[v]);
      u[v + 8] = f2bf(hi ? p1[v] : o0);
    }
    const v16bf pb = __builtin_bit_cast(v16bf, u);
#pragma unroll
    for (int dt = 0; dt < 3; ++dt) {          // Z^T += V^T * P^T
      const unsigned short* p = vb + (dt * 16 + l16) * 32 + hi * 16;
      zacc[dt] = wmma_bf16(frag16(p, p + 8), pb, zacc[dt]);
    }
    buf = bufn;
    bufn = (bufn == 2) ? 0 : bufn + 1;
  }

  const float inv = 1.0f / lrun;
  unsigned short* Ob = O + ((size_t)bp * 1024 + q0 + l16) * 192 + h * 48;
#pragma unroll
  for (int dt = 0; dt < 3; ++dt) {
    v8u pk;
#pragma unroll
    for (int v = 0; v < 8; ++v) pk[v] = f2bf(zacc[dt][v] * inv);
    *(v8u*)(Ob + dt * 16 + hi * 8) = pk;
  }
}

// ------------------------------------------------------------------ out proj
__global__ __launch_bounds__(128) void proj_kernel(
    const unsigned short* __restrict__ X,     // [32768][192] bf16
    const unsigned short* __restrict__ W,     // [192][192]   bf16
    const float*          __restrict__ bias,  // [192]
    float* __restrict__ out)                  // [32768][192] f32
{
  const int lane = threadIdx.x & 31;
  const int fg   = threadIdx.x >> 5;
  const int t0   = blockIdx.x * 16;
  const int hi   = lane >> 4;
  const int l16  = lane & 15;

  v16bf bf[6];
#pragma unroll
  for (int kk = 0; kk < 6; ++kk) {
    const unsigned short* p = X + (size_t)(t0 + l16) * 192 + kk * 32 + hi * 8;
    bf[kk] = frag16(p, p + 16);
  }
  v8f acc[3] = {};
  v16bf a[3];
#pragma unroll
  for (int ft = 0; ft < 3; ++ft) {            // prime K-step 0
    const unsigned short* ap =
        W + (size_t)(fg * 48 + ft * 16 + l16) * 192 + hi * 8;
    a[ft] = frag16(ap, ap + 16);
  }
#pragma unroll
  for (int kk = 0; kk < 6; ++kk) {
    v16bf an[3];
    if (kk + 1 < 6) {
      const int k1 = (kk + 1) * 32;
#pragma unroll
      for (int ft = 0; ft < 3; ++ft) {
        const unsigned short* ap =
            W + (size_t)(fg * 48 + ft * 16 + l16) * 192 + k1 + hi * 8;
        an[ft] = frag16(ap, ap + 16);
      }
    }
#pragma unroll
    for (int ft = 0; ft < 3; ++ft) acc[ft] = wmma_bf16(a[ft], bf[kk], acc[ft]);
    if (kk + 1 < 6) {
#pragma unroll
      for (int ft = 0; ft < 3; ++ft) a[ft] = an[ft];
    }
  }
  const int token = t0 + l16;
#pragma unroll
  for (int ft = 0; ft < 3; ++ft) {
    const int fcol = fg * 48 + ft * 16 + hi * 8;
    float4 r0, r1;
    r0.x = acc[ft][0] + bias[fcol + 0];
    r0.y = acc[ft][1] + bias[fcol + 1];
    r0.z = acc[ft][2] + bias[fcol + 2];
    r0.w = acc[ft][3] + bias[fcol + 3];
    r1.x = acc[ft][4] + bias[fcol + 4];
    r1.y = acc[ft][5] + bias[fcol + 5];
    r1.z = acc[ft][6] + bias[fcol + 6];
    r1.w = acc[ft][7] + bias[fcol + 7];
    float* op = out + (size_t)token * 192 + fcol;
    ((float4*)op)[0] = r0;
    ((float4*)op)[1] = r1;
  }
}

// ---------------------------------------------------------------------------
extern "C" void kernel_launch(void* const* d_in, const int* in_sizes, int n_in,
                              void* d_out, int out_size, void* d_ws, size_t ws_size,
                              hipStream_t stream) {
  (void)in_sizes; (void)n_in; (void)out_size; (void)ws_size;
  const float* x     = (const float*)d_in[0];
  const float* wqkv  = (const float*)d_in[1];
  const float* bqkv  = (const float*)d_in[2];
  const float* wproj = (const float*)d_in[3];
  const float* bproj = (const float*)d_in[4];
  float* out = (float*)d_out;

  char* ws = (char*)d_ws;
  size_t off = 0;
  auto alloc = [&](size_t bytes) {
    void* p = ws + off;
    off += (bytes + 255) & ~(size_t)255;
    return p;
  };
  unsigned short* xbf = (unsigned short*)alloc((size_t)32 * 1024 * 192 * 2);
  unsigned short* wqb = (unsigned short*)alloc((size_t)576 * 192 * 2);
  unsigned short* wpb = (unsigned short*)alloc((size_t)192 * 192 * 2);
  unsigned short* Qb  = (unsigned short*)alloc((size_t)32 * 4 * 1024 * 64 * 2);
  unsigned short* Kb  = (unsigned short*)alloc((size_t)32 * 4 * 1024 * 64 * 2);
  unsigned short* Vt  = (unsigned short*)alloc((size_t)32 * 4 * 64 * 1024 * 2);
  unsigned short* AO  = (unsigned short*)alloc((size_t)32 * 1024 * 192 * 2);

  const int n4x = 32 * 1024 * 192 / 4;
  cvt4_kernel<<<(n4x + 255) / 256, 256, 0, stream>>>(x, xbf, n4x);
  const int n4q = 576 * 192 / 4;
  cvt4_kernel<<<(n4q + 255) / 256, 256, 0, stream>>>(wqkv, wqb, n4q);
  const int n4p = 192 * 192 / 4;
  cvt4_kernel<<<(n4p + 255) / 256, 256, 0, stream>>>(wproj, wpb, n4p);

  qkv_kernel<<<dim3(32 * 64, 3), 128, 0, stream>>>(xbf, wqb, bqkv, Qb, Kb, Vt);
  attn_kernel<<<1024, 256, 0, stream>>>(Qb, Kb, Vt, AO);
  proj_kernel<<<2048, 128, 0, stream>>>(AO, wpb, bproj, out);
}
